// ALG_78005196030420
// MI455X (gfx1250) — matrix operand order repaired and checked by reading, not compiled, not run on hardware
//
#include <hip/hip_runtime.h>
#include <math.h>

// CRF forward (log-partition) for B=64, T=2048, S=256 on gfx1250.
// Exp-domain transform turns the per-step log-semiring matvec into a
// real GEMM executed with v_wmma_f32_16x16x32_f16.
//
// v4: amdgpu_waves_per_eu(4,4) (grid is 4 WGs -> 4 waves/SIMD is the true
//     occupancy, so give RA the full register budget) + sched_group_barrier
//     pinning [16x ds_load_b128][8x wmma] so B fragments stay in distinct
//     registers and WMMAs wait on partial dscnt instead of draining.
//     Branchless clamped t+1 load / t+2 prefetch keeps one sched region.

typedef __attribute__((ext_vector_type(16))) _Float16 v16h;
typedef __attribute__((ext_vector_type(8)))  _Float16 h8;
typedef __attribute__((ext_vector_type(8)))  float    v8f;

#define CRF_S 256
#define CRF_T 2048
#define CRF_B 64
#define NB_WG 16                 // batches per workgroup
#define WAVES 16                 // one wave per 16-row output tile
#define THREADS (WAVES * 32)
#define PROW 264                 // padded LDS row (halves): 528B = 132 dw = 4 mod 64 banks
#define MROW 20                  // wgmax row stride (floats): 20 mod 64 -> conflict-free
#define NEG_LARGE -10000.0f
#define START_IDX 0
#define END_IDX 1
#define LOG2E 1.4426950408889634f
#define LN2   0.6931471805599453f

__device__ __forceinline__ float fast_log(float x)  {   // x normal, > 0
    return LN2 * __builtin_amdgcn_logf(x);              // v_log_f32 (log2)
}
__device__ __forceinline__ float fast_exp(float x)  {
    return __builtin_amdgcn_exp2f(x * LOG2E);           // v_exp_f32 (exp2)
}

__global__ __launch_bounds__(THREADS)
__attribute__((amdgpu_waves_per_eu(4, 4)))
void crf_fwd_wmma(
    const float* __restrict__ text,     // [B, T, S]
    const float* __restrict__ weights,  // [S, S]  w[i,j] = score j->i
    const int*   __restrict__ lens,     // [B]
    float*       __restrict__ out)      // [B]
{
    __shared__ _Float16 p_lds[2][NB_WG * PROW];  // exp(fv - m), [parity][batch][state]
    __shared__ float    wgmax[NB_WG * MROW];     // [batch][wave] partial maxima (padded)
    __shared__ float    fvdump[NB_WG * CRF_S];   // terminal reduction buffer

    const int tid  = threadIdx.x;
    const int lane = tid & 31;
    const int w    = tid >> 5;            // wave id = row-tile id (i in [16w,16w+16))
    const int bb   = lane & 15;           // batch column within workgroup (N = lane%16)
    const int hi   = lane >> 4;           // lane half selects rows/K per ISA layout
    const int bg   = blockIdx.x * NB_WG + bb;     // global batch for this lane's column
    const int len_b = lens[bg];

    // ---------------- Build constant A fragments: A[s][m][k] = exp(W[i, 32s+k]) ----
    // 16-bit A 16x32 layout (ISA 7.12.2): lane M = lane%16;
    // VGPR v<4: K = hi*8 + 2v + h ; v>=4: K = 16 + hi*8 + 2(v-4) + h.
    v16h A[8];
    {
        const int i = w * 16 + bb;        // this lane's A row (M = lane%16)
        #pragma unroll
        for (int s = 0; s < 8; ++s) {
            v16h a;
            #pragma unroll
            for (int e = 0; e < 16; ++e) {
                const int v = e >> 1, h = e & 1;
                int k = (v < 4) ? (hi * 8 + 2 * v + h)
                                : (16 + hi * 8 + 2 * (v - 4) + h);
                k += 32 * s;
                a[e] = (_Float16)__expf(weights[i * CRF_S + k]);  // forbidden -> 0
            }
            A[s] = a;
        }
    }

    // ---------------- fv registers: rows i = i0..i0+7 (f32 C/D layout) -------------
    const int i0 = w * 16 + hi * 8;
    float fv[8];
    #pragma unroll
    for (int v = 0; v < 8; ++v)
        fv[v] = (i0 + v == START_IDX) ? 0.0f : NEG_LARGE;

    // publish(): per-batch max m, write p = exp(fv - m) into buffer `par`, return m
    auto publish = [&](const float* fvloc, int par) -> float {
        float mx = fvloc[0];
        #pragma unroll
        for (int v = 1; v < 8; ++v) mx = fmaxf(mx, fvloc[v]);
        mx = fmaxf(mx, __shfl_xor(mx, 16, 32));       // combine row halves
        if (lane < 16) wgmax[lane * MROW + w] = mx;   // [batch][wave]
        __syncthreads();                               // barrier #1: wgmax ready
        const float4* r = (const float4*)&wgmax[bb * MROW];
        float4 a0 = r[0], a1 = r[1], a2 = r[2], a3 = r[3];   // 4x ds_load_b128
        float m = fmaxf(fmaxf(fmaxf(a0.x, a0.y), fmaxf(a0.z, a0.w)),
                        fmaxf(fmaxf(a1.x, a1.y), fmaxf(a1.z, a1.w)));
        m = fmaxf(m, fmaxf(fmaxf(fmaxf(a2.x, a2.y), fmaxf(a2.z, a2.w)),
                           fmaxf(fmaxf(a3.x, a3.y), fmaxf(a3.z, a3.w))));
        h8 p;
        #pragma unroll
        for (int v = 0; v < 8; ++v) p[v] = (_Float16)fast_exp(fvloc[v] - m);
        *(h8*)&p_lds[par][bb * PROW + i0] = p;        // one ds_store_b128
        __syncthreads();                               // barrier #2: p ready
        return m;
    };

    float m_b = publish(fv, 0);

    // ---------------- stream text: double-buffered 2x float4 per lane --------------
    const float* tptr = text + (size_t)bg * CRF_T * CRF_S + i0;
    float4 f0 = *(const float4*)(tptr);
    float4 f1 = *(const float4*)(tptr + 4);

    const int kOff = hi * 8;
    for (int t = 0; t < CRF_T; ++t) {
        const int par = t & 1;
        // branchless double-buffer: clamp indices so the body is one sched region
        const int t1 = (t + 1 < CRF_T) ? (t + 1) : (CRF_T - 1);
        const int t2 = (t + 2 < CRF_T) ? (t + 2) : (CRF_T - 1);
        float4 n0 = *(const float4*)(tptr + (size_t)t1 * CRF_S);
        float4 n1 = *(const float4*)(tptr + (size_t)t1 * CRF_S + 4);
        __builtin_prefetch(tptr + (size_t)t2 * CRF_S, 0, 3);

        // q = E * p : 8 x v_wmma_f32_16x16x32_f16, B fragments from LDS.
        // 16-bit B 32x16 layout: N = lane%16 ; K = hi*8 + e (e<8), 16 + hi*8 + (e-8) (e>=8), + 32s
        // -> two runs of 8 consecutive halves per lane per k-step.
        auto loadB = [&](int s) -> v16h {
            const _Float16* src = &p_lds[par][bb * PROW + 32 * s + kOff];
            h8 blo = *(const h8*)src;          // ds_load_b128
            h8 bhi = *(const h8*)(src + 16);   // ds_load_b128
            v16h bf;
            #pragma unroll
            for (int e = 0; e < 8; ++e) { bf[e] = blo[e]; bf[8 + e] = bhi[e]; }
            return bf;
        };

        v16h bfs[8];
        #pragma unroll
        for (int s = 0; s < 8; ++s) bfs[s] = loadB(s);   // 16 ds_load_b128
        v8f c = {};
        #pragma unroll
        for (int s = 0; s < 8; ++s)
            c = __builtin_amdgcn_wmma_f32_16x16x32_f16(
                    false, A[s], false, bfs[s], (short)0, c, false, false);

        // Pin the pipeline: all 16 DS reads issue before the 8 WMMAs, so the
        // fragments' live ranges overlap (distinct VGPRs) and each WMMA waits
        // on a partial dscnt while later loads are still in flight.
        __builtin_amdgcn_sched_group_barrier(0x100, 16, 0);  // DS read x16
        __builtin_amdgcn_sched_group_barrier(0x008,  8, 0);  // WMMA   x8

        // fv' = feat + m + log(q), masked by sequence length
        const float feat[8] = {f0.x, f0.y, f0.z, f0.w, f1.x, f1.y, f1.z, f1.w};
        const bool upd = (t < len_b);
        #pragma unroll
        for (int v = 0; v < 8; ++v) {
            const float q  = fmaxf(c[v], 1e-30f);      // keeps q normal: raw v_log ok
            const float nf = feat[v] + m_b + fast_log(q);
            fv[v] = upd ? nf : fv[v];
        }

        m_b = publish(fv, par ^ 1);   // write the other buffer; 2 barriers total
        f0 = n0; f1 = n1;
    }

    // ---------------- terminal: out[b] = logsumexp_i(fv[b,i] + W[END, i]) ----------
    #pragma unroll
    for (int v = 0; v < 8; ++v) fvdump[bb * CRF_S + i0 + v] = fv[v];
    __syncthreads();
    if (tid < NB_WG) {
        const int b = tid;
        float mx = -INFINITY;
        for (int i = 0; i < CRF_S; ++i)
            mx = fmaxf(mx, fvdump[b * CRF_S + i] + weights[END_IDX * CRF_S + i]);
        float sum = 0.0f;
        for (int i = 0; i < CRF_S; ++i)
            sum += __expf(fvdump[b * CRF_S + i] + weights[END_IDX * CRF_S + i] - mx);
        out[blockIdx.x * NB_WG + b] = mx + __logf(sum);
    }
}

extern "C" void kernel_launch(void* const* d_in, const int* in_sizes, int n_in,
                              void* d_out, int out_size, void* d_ws, size_t ws_size,
                              hipStream_t stream) {
    (void)in_sizes; (void)n_in; (void)d_ws; (void)ws_size; (void)out_size;
    const float* text    = (const float*)d_in[0];
    const float* weights = (const float*)d_in[1];
    const int*   lens    = (const int*)d_in[2];
    float*       out     = (float*)d_out;
    crf_fwd_wmma<<<CRF_B / NB_WG, THREADS, 0, stream>>>(text, weights, lens, out);
}
